// HamNaiveDynMessage_24953759989871
// MI455X (gfx1250) — matrix operand order repaired and checked by reading, not compiled, not run on hardware
//
#include <hip/hip_runtime.h>
#include <cfloat>
#include <math.h>

#define F_DIM 128
#define UNITS 128

// padded fragment chunk: 16 bf16 (32B) + 16B pad -> 48B lane stride (bank-conflict friendly)
#define CHUNK_BYTES 48
#define WATT_FRAGS  (4 * 8)      // kb * nb
#define WE_FRAGS    (16 * 8)
#define WATT_BYTES  ((size_t)WATT_FRAGS * 32 * CHUNK_BYTES)   //  48 KB
#define WE_BYTES    ((size_t)WE_FRAGS   * 32 * CHUNK_BYTES)   // 192 KB
#define SMEM_BYTES  (WATT_BYTES + WE_BYTES)                   // 240 KB (<320KB/WGP)

typedef __attribute__((ext_vector_type(16))) __bf16 v16bf;
typedef __attribute__((ext_vector_type(8)))  __bf16 v8bf;
typedef __attribute__((ext_vector_type(8)))  float  v8f;
typedef __attribute__((ext_vector_type(4)))  unsigned int v4u;
typedef __attribute__((ext_vector_type(8)))  unsigned int v8u;

// ---------------------------------------------------------------- helpers
__device__ __forceinline__ void atomicMaxF(float* addr, float v) {
    if (v >= 0.0f) atomicMax((int*)addr, __float_as_int(v));
    else           atomicMin((unsigned int*)addr, __float_as_uint(v));
}

__device__ __forceinline__ v8f wmma_bf16(v16bf a, v16bf b, v8f c) {
    return __builtin_amdgcn_wmma_f32_16x16x32_bf16(
        /*neg_a=*/false, a, /*neg_b=*/false, b,
        /*c_mod=*/(short)0, c, /*reuse_a=*/false, /*reuse_b=*/false);
}

// Tensor Data Mover: flat 1-D copy of `ndwords` dwords from global `gptr` to
// LDS byte offset `lds_off`.  D# per CDNA5 ISA (groups 0/1,
// groups 2/3 NULL -> 2-operand tensor_load_to_lds form).  Tracked by TENSORcnt.
__device__ __forceinline__ void tdm_load_to_lds(unsigned lds_off, const void* gptr,
                                                unsigned ndwords) {
    unsigned long long ga = (unsigned long long)gptr;
    v4u g0;
    g0[0] = 1u;                                        // count=1 (valid), user mode
    g0[1] = lds_off;                                   // lds_addr (bytes)
    g0[2] = (unsigned)(ga & 0xffffffffu);              // global_addr[31:0]
    g0[3] = (unsigned)((ga >> 32) & 0x01ffffffu)       // global_addr[56:32]
          | (2u << 30);                                // type = 2 ("image")
    v8u g1;
    g1[0] = (2u << 16);                                // data_size=4B; mask=0; no pad
    g1[1] = (ndwords & 0xffffu) << 16;                 // tensor_dim0[15:0] @ bit48
    g1[2] = ((ndwords >> 16) & 0xffffu)                // tensor_dim0[31:16]
          | (1u << 16);                                // tensor_dim1 = 1
    g1[3] = (ndwords & 0xffffu) << 16;                 // tile_dim0 @ bits 127:112
    g1[4] = 0;                                         // tile_dim1/2 unused
    g1[5] = ndwords;                                   // tensor_dim0_stride[31:0]
    g1[6] = 0;
    g1[7] = 0;
    asm volatile("tensor_load_to_lds %0, %1" :: "s"(g0), "s"(g1) : "memory");
}

// A-fragment (16x32 bf16, ISA layout): lane holds row m=lane&15.
// lanes 0-15: slots 0..7 -> K=kb*32+0..7,  slots 8..15 -> K=kb*32+16..23
// lanes 16-31: slots 0..7 -> K=kb*32+8..15, slots 8..15 -> K=kb*32+24..31
__device__ __forceinline__ v16bf load_a_row(const float* __restrict__ row, int kb, int half) {
    const float* s0 = row + kb * 32 + half * 8;
    v16bf a;
#pragma unroll
    for (int j = 0; j < 8; ++j) a[j]     = (__bf16)s0[j];
#pragma unroll
    for (int j = 0; j < 8; ++j) a[8 + j] = (__bf16)s0[16 + j];
    return a;
}

__device__ __forceinline__ v16bf load_a_diff(const float* __restrict__ rv,
                                             const float* __restrict__ ru, int kb, int half) {
    const float* v0 = rv + kb * 32 + half * 8;
    const float* u0 = ru + kb * 32 + half * 8;
    v16bf a;
#pragma unroll
    for (int j = 0; j < 8; ++j) a[j]     = (__bf16)(v0[j] - u0[j]);
#pragma unroll
    for (int j = 0; j < 8; ++j) a[8 + j] = (__bf16)(v0[16 + j] - u0[16 + j]);
    return a;
}

// B fragment from LDS (padded 48B/lane chunks -> two ds_load_b128, low conflict)
__device__ __forceinline__ v16bf load_b_frag_lds(const char* base, int frag, int lane) {
    const char* p = base + (size_t)(frag * 32 + lane) * CHUNK_BYTES;
    v8bf lo = *(const v8bf*)p;
    v8bf hi = *(const v8bf*)(p + 16);
    v16bf r;
#pragma unroll
    for (int j = 0; j < 8; ++j) { r[j] = lo[j]; r[8 + j] = hi[j]; }
    return r;
}

// ---------------------------------------------------------------- kernels
__global__ void init_kernel(float* amax, float* denom, float* mv, int n_nodes, int total) {
    int i = blockIdx.x * blockDim.x + threadIdx.x;
    if (i < n_nodes) { amax[i] = -FLT_MAX; denom[i] = 0.0f; }
    if (i < total) mv[i] = 0.0f;
}

// Repack W (K x N, f32 row-major) into padded bf16 WMMA B-fragments.
// frag = kb*NB + nb ; lane L<16 -> N = nb*16+L, K = kb*32 + {0..7, 16..23}
//                     lane L>=16 -> N = nb*16+(L-16), K = kb*32 + {8..15, 24..31}
__global__ void wfrag_kernel(const float* __restrict__ W, char* __restrict__ out, int K, int N) {
    int t = blockIdx.x * blockDim.x + threadIdx.x;
    int NB = N / 16;
    int total = (K / 32) * NB * 32;
    if (t >= total) return;
    int lane = t & 31;
    int frag = t >> 5;
    int nb = frag % NB;
    int kb = frag / NB;
    int n = nb * 16 + (lane & 15);
    int kbase = kb * 32 + ((lane < 16) ? 0 : 8);
    __bf16* dst = (__bf16*)(out + (size_t)t * CHUNK_BYTES);
#pragma unroll
    for (int j = 0; j < 16; ++j) dst[j] = (__bf16)W[(size_t)(kbase + (j & 7) + ((j >> 3) << 4)) * N + n];
}

// One wave per edge: a[e] = [p_uv | q_uv | he] . W_align + b ; atomic segment max.
__global__ void align_kernel(const float* __restrict__ p, const float* __restrict__ q,
                             const float* __restrict__ he,
                             const long long* __restrict__ ui, const long long* __restrict__ vi,
                             const float* __restrict__ W_align, const float* __restrict__ b_align,
                             float* __restrict__ a_out, float* __restrict__ amax, int n_edges) {
    int gtid = blockIdx.x * blockDim.x + threadIdx.x;
    int e = gtid >> 5;
    int lane = gtid & 31;
    if (e >= n_edges) return;
    long long u = ui[e], v = vi[e];
    const float* pu = p + u * (long long)F_DIM;
    const float* pv = p + v * (long long)F_DIM;
    const float* qu = q + u * (long long)F_DIM;
    const float* qv = q + v * (long long)F_DIM;
    const float* hee = he + (long long)e * F_DIM;
    float s = 0.0f;
#pragma unroll
    for (int t = 0; t < F_DIM / 32; ++t) {
        int i = lane + t * 32;
        s += (pv[i] - pu[i]) * W_align[i];
        s += (qv[i] - qu[i]) * W_align[F_DIM + i];
        s += hee[i] * W_align[2 * F_DIM + i];
    }
#pragma unroll
    for (int off = 16; off > 0; off >>= 1) s += __shfl_xor(s, off, 32);
    if (lane == 0) {
        float a = s + b_align[0];
        a_out[e] = a;
        atomicMaxF(&amax[u], a);
    }
}

__global__ void ex_kernel(const float* __restrict__ a, const long long* __restrict__ ui,
                          const float* __restrict__ amax, float* __restrict__ exv,
                          float* __restrict__ denom, int n_edges) {
    int e = blockIdx.x * blockDim.x + threadIdx.x;
    if (e >= n_edges) return;
    long long u = ui[e];
    float am = amax[u];
    if (am == -FLT_MAX) am = 0.0f;
    float v = expf(a[e] - am);
    exv[e] = v;
    atomicAdd(&denom[u], v);
}

// Persistent waves; weights TDM-staged into LDS once per block; wave handles a
// 16-edge tile per iteration: attend GEMM (16x128x128) + me GEMM (16x512x128).
__global__ void __launch_bounds__(256, 1)
edge_main_kernel(const float* __restrict__ hv, const float* __restrict__ p,
                 const float* __restrict__ q,
                 const long long* __restrict__ ui, const long long* __restrict__ vi,
                 const char* __restrict__ WattG, const float* __restrict__ b_attend,
                 const char* __restrict__ WeG, const float* __restrict__ b_e,
                 const float* __restrict__ exv, const float* __restrict__ denom,
                 float* __restrict__ mv, float* __restrict__ me_out, int n_edges) {
    extern __shared__ char smem[];
    char* sWatt = smem;
    char* sWe   = smem + WATT_BYTES;

    // Wave 0 stages both weight-fragment blobs via the Tensor Data Mover.
    if (threadIdx.x < 32) {
        tdm_load_to_lds(0u, WattG, (unsigned)(WATT_BYTES / 4));
        tdm_load_to_lds((unsigned)WATT_BYTES, WeG, (unsigned)(WE_BYTES / 4));
        __builtin_amdgcn_s_wait_tensorcnt(0);
    }
    __syncthreads();

    int lane = threadIdx.x & 31;
    int half = lane >> 4;
    int mrow = lane & 15;
    int gwave = blockIdx.x * (blockDim.x >> 5) + (threadIdx.x >> 5);
    int nwaves = gridDim.x * (blockDim.x >> 5);
    int ntiles = (n_edges + 15) >> 4;

    for (int tile = gwave; tile < ntiles; tile += nwaves) {   // wave-uniform loop
        int base = tile * 16;

        // A-matrix row for this lane (both halves hold the same edge row, different K)
        int e_a = base + mrow; if (e_a >= n_edges) e_a = n_edges - 1;
        long long va = vi[e_a], ua = ui[e_a];
        const float* hvV = hv + va * (long long)F_DIM;
        const float* hvU = hv + ua * (long long)F_DIM;
        const float* pV = p + va * (long long)F_DIM;
        const float* pU = p + ua * (long long)F_DIM;
        const float* qV = q + va * (long long)F_DIM;
        const float* qU = q + ua * (long long)F_DIM;

        // C-row metadata: VGPR r of this lane is edge m = half*8 + r
        long long u_r[8];
        float alpha_r[8];
        int e_r[8];
#pragma unroll
        for (int r = 0; r < 8; ++r) {
            int e = base + half * 8 + r;
            e_r[r] = e;
            int ec = (e < n_edges) ? e : n_edges - 1;
            long long u = ui[ec];
            u_r[r] = u;
            alpha_r[r] = exv[ec] / fmaxf(denom[u], 1e-16f);
        }

        // ---- attend = lrelu(hv_v @ W_attend + b) ; mv += alpha * attend -------
        v8f ca[8];
#pragma unroll
        for (int nb = 0; nb < 8; ++nb) ca[nb] = (v8f){};
#pragma unroll
        for (int kb = 0; kb < 4; ++kb) {
            v16bf a = load_a_row(hvV, kb, half);
#pragma unroll
            for (int nb = 0; nb < 8; ++nb) {
                v16bf b = load_b_frag_lds(sWatt, kb * 8 + nb, lane);
                ca[nb] = wmma_bf16(a, b, ca[nb]);
            }
        }
#pragma unroll
        for (int nb = 0; nb < 8; ++nb) {
            int n = nb * 16 + mrow;
            float bn = b_attend[n];
#pragma unroll
            for (int r = 0; r < 8; ++r) {
                if (e_r[r] < n_edges) {
                    float t = ca[nb][r] + bn;
                    t = (t > 0.0f) ? t : 0.2f * t;
                    atomicAdd(&mv[u_r[r] * UNITS + n], alpha_r[r] * t);
                }
            }
        }

        // ---- me = lrelu([hv_u | p_uv | q_uv | hv_v] @ W_e + b_e) -------------
        v8f cm[8];
#pragma unroll
        for (int nb = 0; nb < 8; ++nb) cm[nb] = (v8f){};
#pragma unroll
        for (int kb = 0; kb < 16; ++kb) {
            v16bf a;
            if (kb < 4)       a = load_a_row (hvU, kb,      half);
            else if (kb < 8)  a = load_a_diff(pV, pU, kb-4, half);
            else if (kb < 12) a = load_a_diff(qV, qU, kb-8, half);
            else              a = load_a_row (hvV, kb-12,   half);
#pragma unroll
            for (int nb = 0; nb < 8; ++nb) {
                v16bf b = load_b_frag_lds(sWe, kb * 8 + nb, lane);
                cm[nb] = wmma_bf16(a, b, cm[nb]);
            }
        }
#pragma unroll
        for (int nb = 0; nb < 8; ++nb) {
            int n = nb * 16 + mrow;
            float bn = b_e[n];
#pragma unroll
            for (int r = 0; r < 8; ++r) {
                if (e_r[r] < n_edges) {
                    float t = cm[nb][r] + bn;
                    me_out[(long long)e_r[r] * UNITS + n] = (t > 0.0f) ? t : 0.2f * t;
                }
            }
        }
    }
}

__global__ void elu_kernel(const float* __restrict__ mv, float* __restrict__ out, int total) {
    int i = blockIdx.x * blockDim.x + threadIdx.x;
    if (i >= total) return;
    float v = mv[i];
    out[i] = (v > 0.0f) ? v : expm1f(v);
}

// ---------------------------------------------------------------- launch
extern "C" void kernel_launch(void* const* d_in, const int* in_sizes, int n_in,
                              void* d_out, int out_size, void* d_ws, size_t ws_size,
                              hipStream_t stream) {
    const float* hv        = (const float*)d_in[0];
    const float* he        = (const float*)d_in[1];
    const float* p         = (const float*)d_in[2];
    const float* q         = (const float*)d_in[3];
    const long long* eidx  = (const long long*)d_in[4];   // int64 (2, M)
    const float* W_attend  = (const float*)d_in[5];
    const float* b_attend  = (const float*)d_in[6];
    const float* W_align   = (const float*)d_in[7];
    const float* b_align   = (const float*)d_in[8];
    const float* W_e       = (const float*)d_in[9];
    const float* b_e       = (const float*)d_in[10];

    int n_nodes = in_sizes[0] / F_DIM;
    int n_edges = in_sizes[1] / F_DIM;
    const long long* ui = eidx;
    const long long* vi = eidx + n_edges;

    // workspace partition (256B aligned)
    char* wp = (char*)d_ws;
    auto alloc = [&](size_t bytes) {
        char* r = wp; wp += (bytes + 255) & ~(size_t)255; return r;
    };
    float* amax  = (float*)alloc((size_t)n_nodes * 4);
    float* denom = (float*)alloc((size_t)n_nodes * 4);
    float* a_sc  = (float*)alloc((size_t)n_edges * 4);
    float* exv   = (float*)alloc((size_t)n_edges * 4);
    float* mv    = (float*)alloc((size_t)n_nodes * UNITS * 4);
    char*  WattG = (char*)alloc(WATT_BYTES);   //  48 KB padded fragments
    char*  WeG   = (char*)alloc(WE_BYTES);     // 192 KB padded fragments

    float* out_mv = (float*)d_out;
    float* out_me = out_mv + (size_t)n_nodes * UNITS;

    static bool attr_set = false;  // idempotent, deterministic
    if (!attr_set) {
        hipFuncSetAttribute(reinterpret_cast<const void*>(edge_main_kernel),
                            hipFuncAttributeMaxDynamicSharedMemorySize,
                            (int)SMEM_BYTES);
        attr_set = true;
    }

    int mv_total = n_nodes * UNITS;
    init_kernel<<<(mv_total + 255) / 256, 256, 0, stream>>>(amax, denom, mv, n_nodes, mv_total);

    wfrag_kernel<<<(WATT_FRAGS * 32 + 255) / 256, 256, 0, stream>>>(W_attend, WattG, 128, 128);
    wfrag_kernel<<<(WE_FRAGS * 32 + 255) / 256, 256, 0, stream>>>(W_e, WeG, 512, 128);

    long long athreads = (long long)n_edges * 32;
    align_kernel<<<(unsigned)((athreads + 255) / 256), 256, 0, stream>>>(
        p, q, he, ui, vi, W_align, b_align, a_sc, amax, n_edges);

    ex_kernel<<<(n_edges + 255) / 256, 256, 0, stream>>>(a_sc, ui, amax, exv, denom, n_edges);

    int ntiles = (n_edges + 15) / 16;
    int blocks = (ntiles + 7) / 8;
    if (blocks > 1024) blocks = 1024;
    edge_main_kernel<<<blocks, 256, SMEM_BYTES, stream>>>(
        hv, p, q, ui, vi, WattG, b_attend, WeG, b_e, exv, denom, mv, out_me, n_edges);

    elu_kernel<<<(mv_total + 255) / 256, 256, 0, stream>>>(mv, out_mv, mv_total);
}
